// LandmarkMemory_23304492548364
// MI455X (gfx1250) — matrix operand order repaired and checked by reading, not compiled, not run on hardware
//
#include <hip/hip_runtime.h>
#include <hip/hip_bf16.h>
#include <math.h>

// LandmarkMemory: N=2048, K=128, H=512, TOPK=8. Memory-bound (~553MB -> ~24us
// @23.3TB/s). One workgroup per n; sim via v_wmma_f32_16x16x32_f16 with q
// broadcast into all 16 B-columns (batched GEMV on the tensor path).
// Async DMA (global_load_async_to_lds_b128) staging pipelined against WMMA
// via staged s_wait_asynccnt immediates (async loads complete in order).

#define N_ 2048
#define K_ 128
#define H_ 512

typedef __attribute__((ext_vector_type(16))) _Float16 v16h;
typedef __attribute__((ext_vector_type(8)))  float    v8f;

union V16 { v16h v; _Float16 h[16]; uint4 u[2]; };
union V8  { v8f  v; float f[8]; };

#if defined(__has_builtin)
#  if __has_builtin(__builtin_amdgcn_global_load_async_to_lds_b128) && \
      __has_builtin(__builtin_amdgcn_s_wait_asynccnt)
#    define USE_ASYNC_LDS 1
#  endif
#endif

#ifdef USE_ASYNC_LDS
// Param types per hipcc diagnostic: (v4i addrspace(1)*, v4i addrspace(3)*, Ii, Ii)
typedef int v4i_t __attribute__((__vector_size__(4 * sizeof(int))));
typedef __attribute__((address_space(1))) v4i_t* g_v4i_p;
typedef __attribute__((address_space(3))) v4i_t* l_v4i_p;
#endif

// LDS layout in float slots
#define L_QSH16 0                    // 512 f16 (256 float slots): q row, f16
#define L_DOTS  256                  // 128 f32: raw dots
#define L_SSQ   384                  // 128 f32: row sum-of-squares
#define L_QPART 512                  // 8 f32: per-wave |q|^2 partials
#define L_ATTN  520                  // 8 f32: softmax weights
#define L_IDX   528                  // 8 i32: top-k indices
#define L_STAGE 544                  // 8 waves * 16 rows * 512 f32 (16B aligned)
#define SMEM_FLOATS (L_STAGE + 8 * 16 * H_)

// One 16x16x32 WMMA chunk. C is the chunk index (h0 = 32*C); the wait
// immediate releases the chunk once the in-order async DMA has delivered
// h < 32*(C+1) for all 16 rows (4 completions per chunk, 64 total).
template <int C>
__device__ __forceinline__ void do_chunk(const float* __restrict__ rowp,
                                         const _Float16* __restrict__ qh,
                                         int abase, int qb,
                                         float& ss, V8& cacc)
{
#ifdef USE_ASYNC_LDS
    __builtin_amdgcn_s_wait_asynccnt(60 - 4 * C);
#endif
    const int h0 = C * 32;
    float fa[16];
    *(float4*)&fa[0]  = *(const float4*)(rowp + h0 + abase);
    *(float4*)&fa[4]  = *(const float4*)(rowp + h0 + abase + 4);
    *(float4*)&fa[8]  = *(const float4*)(rowp + h0 + 16 + abase);
    *(float4*)&fa[12] = *(const float4*)(rowp + h0 + 16 + abase + 4);
    V16 av, bv;
    bv.u[0] = *(const uint4*)(qh + h0 + qb);
    bv.u[1] = *(const uint4*)(qh + h0 + qb + 16);
    #pragma unroll
    for (int j = 0; j < 16; ++j) {
        ss += fa[j] * fa[j];
        av.h[j] = (_Float16)fa[j];
    }
    cacc.v = __builtin_amdgcn_wmma_f32_16x16x32_f16(
        false, av.v, false, bv.v, (short)0, cacc.v, false, false);
}

__global__ __launch_bounds__(256)
void lm_kernel(const float* __restrict__ q,
               const float* __restrict__ mem_k,
               const float* __restrict__ mem_v,
               const float* __restrict__ mem_m,
               float* __restrict__ out)
{
    extern __shared__ float smem[];
    const int n    = blockIdx.x;
    const int tid  = threadIdx.x;
    const int lane = tid & 31;
    const int wave = tid >> 5;

    _Float16* qh  = (_Float16*)(smem + L_QSH16);
    float* dots   = smem + L_DOTS;
    float* ssq    = smem + L_SSQ;
    float* qpart  = smem + L_QPART;
    float* attnw  = smem + L_ATTN;
    int*   idxs   = (int*)(smem + L_IDX);
    float* stage  = smem + L_STAGE + wave * (16 * H_);

    const int row = lane & 15;    // A-matrix row owned by this lane (M = lane%16)
    const int hi  = lane >> 4;    // lane half selects the K-subrange of A

    const float* krows = mem_k + ((size_t)n * K_ + (size_t)wave * 16) * H_;

#ifdef USE_ASYNC_LDS
    {   // DMA 16 rows (32KB) to this wave's staging buffer. Lane l copies 16B
        // segments of row (l&15); each completed instruction advances every
        // row by 32B, so data arrives front-to-back across all rows.
        const char* gbase = (const char*)(krows + (size_t)row * H_) + hi * 16;
        char*       lbase = (char*)(stage + row * H_) + hi * 16;
        #pragma unroll 8
        for (int i = 0; i < 64; ++i) {
            __builtin_amdgcn_global_load_async_to_lds_b128(
                (g_v4i_p)(gbase + i * 32),
                (l_v4i_p)(lbase + i * 32),
                0, 0);
        }
    }
#endif

    // ---- stage q (as f16) to LDS, reduce |q|^2 ---------------------------
    {
        float2 qv = *(const float2*)(q + (size_t)n * H_ + tid * 2);
        qh[tid * 2]     = (_Float16)qv.x;
        qh[tid * 2 + 1] = (_Float16)qv.y;
        float ps = qv.x * qv.x + qv.y * qv.y;
        #pragma unroll
        for (int off = 16; off >= 1; off >>= 1) ps += __shfl_xor(ps, off, 32);
        if (lane == 0) qpart[wave] = ps;
    }
    __syncthreads();

#ifndef USE_ASYNC_LDS
    {   // fallback: coalesced float4 global->reg->LDS staging
        #pragma unroll 4
        for (int i = 0; i < 64; ++i) {
            int f4 = i * 32 + lane;        // float4 index within 16x512 tile
            int r  = f4 >> 7;              // 128 float4 per row
            int c4 = f4 & 127;
            float4 v = *(const float4*)(krows + (size_t)r * H_ + (size_t)c4 * 4);
            *(float4*)(stage + r * H_ + c4 * 4) = v;
        }
    }
#endif

    // ---- per-wave: 16 k-rows -> dots (WMMA) + sum-of-squares -------------
    // 16-bit A 16x32 layout: lanes<16 hold K-offsets {0..7,16..23},
    // lanes>=16 hold {8..15,24..31}, 2 packed per VGPR.
    // B 32x16: lanes<16 rows K={0..7,16..23}, lanes>=16 rows K={8..15,24..31}; all 16 columns
    // carry the same q chunk (broadcast GEMV).
    V8 cacc;
    #pragma unroll
    for (int j = 0; j < 8; ++j) cacc.f[j] = 0.f;
    float ss = 0.f;
    const float* rowp  = stage + row * H_;
    const int    abase = hi * 8;
    const int    qb    = hi * 8;

    do_chunk<0>(rowp, qh, abase, qb, ss, cacc);
    do_chunk<1>(rowp, qh, abase, qb, ss, cacc);
    do_chunk<2>(rowp, qh, abase, qb, ss, cacc);
    do_chunk<3>(rowp, qh, abase, qb, ss, cacc);
    do_chunk<4>(rowp, qh, abase, qb, ss, cacc);
    do_chunk<5>(rowp, qh, abase, qb, ss, cacc);
    do_chunk<6>(rowp, qh, abase, qb, ss, cacc);
    do_chunk<7>(rowp, qh, abase, qb, ss, cacc);
    do_chunk<8>(rowp, qh, abase, qb, ss, cacc);
    do_chunk<9>(rowp, qh, abase, qb, ss, cacc);
    do_chunk<10>(rowp, qh, abase, qb, ss, cacc);
    do_chunk<11>(rowp, qh, abase, qb, ss, cacc);
    do_chunk<12>(rowp, qh, abase, qb, ss, cacc);
    do_chunk<13>(rowp, qh, abase, qb, ss, cacc);
    do_chunk<14>(rowp, qh, abase, qb, ss, cacc);
    do_chunk<15>(rowp, qh, abase, qb, ss, cacc);

    ss += __shfl_xor(ss, 16, 32);               // combine the two row halves
    if (lane < 16) ssq[wave * 16 + row] = ss;
    // C layout: lane j holds column j; VGPR r = row r (lanes 0-15) / r+8 (16-31)
    if (lane == 0) {
        #pragma unroll
        for (int j = 0; j < 8; ++j) dots[wave * 16 + j] = cacc.f[j];
    }
    if (lane == 16) {
        #pragma unroll
        for (int j = 0; j < 8; ++j) dots[wave * 16 + 8 + j] = cacc.f[j];
    }
    __syncthreads();

    // ---- wave 0: cosine sims, mask, top-8, softmax -----------------------
    if (wave == 0) {
        float qss = 0.f;
        #pragma unroll
        for (int w = 0; w < 8; ++w) qss += qpart[w];
        const float qd = fmaxf(sqrtf(qss), 1e-12f);
        const float* mrow = mem_m + (size_t)n * K_;
        const int kb = lane * 4;
        float v[4];
        #pragma unroll
        for (int j = 0; j < 4; ++j) {
            int k    = kb + j;
            float kd = fmaxf(sqrtf(ssq[k]), 1e-12f);
            float s  = dots[k] / (qd * kd);
            if (mrow[k] <= 0.f) s = -1e9f;
            v[j] = s;
        }
        float topv[8]; int topi[8];
        #pragma unroll
        for (int t = 0; t < 8; ++t) {
            float m = v[0]; int mi = kb;
            #pragma unroll
            for (int j = 1; j < 4; ++j)
                if (v[j] > m) { m = v[j]; mi = kb + j; }
            #pragma unroll
            for (int off = 16; off >= 1; off >>= 1) {
                float om = __shfl_xor(m, off, 32);
                int   oi = __shfl_xor(mi, off, 32);
                if (om > m || (om == m && oi < mi)) { m = om; mi = oi; }
            }
            topv[t] = m; topi[t] = mi;
            if ((mi >> 2) == lane) v[mi & 3] = -INFINITY;   // lowest-index ties
        }
        const float mx = topv[0];
        float e[8], sum = 0.f;
        #pragma unroll
        for (int t = 0; t < 8; ++t) { e[t] = __expf(topv[t] - mx); sum += e[t]; }
        if (lane == 0) {
            float inv = 1.f / sum;
            #pragma unroll
            for (int t = 0; t < 8; ++t) { attnw[t] = e[t] * inv; idxs[t] = topi[t]; }
        }
    }
    __syncthreads();

    // ---- weighted gather of selected v rows ------------------------------
    {
        const int h = tid * 2;
        float2 acc; acc.x = 0.f; acc.y = 0.f;
        #pragma unroll
        for (int t = 0; t < 8; ++t) {
            const float a = attnw[t];
            const float* vrow = mem_v + ((size_t)n * K_ + idxs[t]) * H_;
            float2 vv = *(const float2*)(vrow + h);
            acc.x += a * vv.x;
            acc.y += a * vv.y;
        }
        *(float2*)(out + (size_t)n * H_ + h) = acc;
    }
}

extern "C" void kernel_launch(void* const* d_in, const int* in_sizes, int n_in,
                              void* d_out, int out_size, void* d_ws, size_t ws_size,
                              hipStream_t stream) {
    (void)in_sizes; (void)n_in; (void)out_size; (void)d_ws; (void)ws_size;
    const float* q     = (const float*)d_in[0];
    const float* mem_k = (const float*)d_in[1];
    const float* mem_v = (const float*)d_in[2];
    const float* mem_m = (const float*)d_in[3];
    float* out = (float*)d_out;
    const size_t shbytes = (size_t)SMEM_FLOATS * sizeof(float);
    lm_kernel<<<dim3(N_), dim3(256), shbytes, stream>>>(q, mem_k, mem_v, mem_m, out);
}
